// ConfSMoEModel_82231443849382
// MI455X (gfx1250) — compile-verified, matrix operand order checked by reading, not run on hardware
//
#include <hip/hip_runtime.h>
#include <math.h>

#define B_SZ   4096
#define CSV    2048
#define EMB    256
#define FUS    768
#define NE     8
#define NC     12
#define BN_INV 0.9999950000374997f   // 1/sqrt(1 + 1e-5)

typedef _Float16 half_t;
typedef __attribute__((ext_vector_type(16))) _Float16 v16h;
typedef __attribute__((ext_vector_type(8)))  _Float16 v8h;
typedef __attribute__((ext_vector_type(8)))  float    v8f;
typedef __attribute__((ext_vector_type(4)))  unsigned int u32x4;
typedef __attribute__((ext_vector_type(8)))  unsigned int u32x8;

// ---------------------------------------------------------------------------
// WMMA helpers: 16-bit A/B fragment layout (CDNA5 ISA 7.12.2):
//   lane 0-15 : row = lane,    kb=0 -> K = 0..7 and 16..23
//   lane 16-31: row = lane-16, kb=8 -> K = 8..15 and 24..31
// With row-major-in-K storage both halves are contiguous 16B loads.
// ---------------------------------------------------------------------------
static __device__ __forceinline__ v16h frag_load(const half_t* __restrict__ p, int kb) {
  v8h lo = *(const v8h*)(p + kb);
  v8h hi = *(const v8h*)(p + 16 + kb);
  v16h r;
#pragma unroll
  for (int i = 0; i < 8; ++i) { r[i] = lo[i]; r[8 + i] = hi[i]; }
  return r;
}

static __device__ __forceinline__ v8f wmma16(v16h a, v16h b, v8f c) {
  return __builtin_amdgcn_wmma_f32_16x16x32_f16(false, a, false, b, (short)0, c,
                                                false, false);
}

// ---------------------------------------------------------------------------
// Tensor Data Mover: 2D tile load Global -> LDS (CDNA5 ISA ch.8 D# layout).
// Loads tileN rows x tileK f16 elements (row stride strideK elems) into LDS
// packed as [row][k] (row stride tileK elems). Wave-level op; EXEC ignored.
// Two-group form (VADDR2/3 = NULL -> up to 2D tensors).
// ---------------------------------------------------------------------------
static __device__ __forceinline__ void tdm_load_tile_f16(
    const half_t* gsrc, unsigned ldsAddr, unsigned tileK, unsigned tileN,
    unsigned strideK) {
  unsigned long long ga = (unsigned long long)gsrc;
  u32x4 g0;
  g0[0] = 1u;                                              // count=1, user mode
  g0[1] = ldsAddr;                                         // lds_addr
  g0[2] = (unsigned)(ga & 0xFFFFFFFFull);                  // global_addr[31:0]
  g0[3] = (unsigned)((ga >> 32) & 0x01FFFFFFull) | (2u << 30); // [56:32] | type=2
  u32x8 g1;
  unsigned long long s0 = (unsigned long long)strideK;     // tensor_dim0_stride
  unsigned td0 = strideK;                                  // tensor_dim0 >= tileK
  unsigned td1 = tileN;                                    // tensor_dim1
  g1[0] = (1u << 16);                                      // data_size=1 (2 bytes)
  g1[1] = (td0 & 0xFFFFu) << 16;                           // dim0[15:0] @ [31:16]
  g1[2] = (td0 >> 16) | ((td1 & 0xFFFFu) << 16);           // dim0[31:16] | dim1[15:0]
  g1[3] = (td1 >> 16) | (tileK << 16);                     // dim1[31:16] | tile_dim0
  g1[4] = tileN;                                           // tile_dim1 (tile_dim2=0)
  g1[5] = (unsigned)(s0 & 0xFFFFFFFFull);                  // dim0_stride[31:0]
  g1[6] = (unsigned)((s0 >> 32) & 0xFFFFu);                // dim0_stride[47:32]
  g1[7] = 0u;                                              // dim1_stride hi (unused)
  asm volatile("tensor_load_to_lds %0, %1" :: "s"(g0), "s"(g1) : "memory");
}

// ---------------------------------------------------------------------------
// Conversion kernels
// ---------------------------------------------------------------------------
__global__ void k_cvt16(const float* __restrict__ src, half_t* __restrict__ dst, int n) {
  int i = blockIdx.x * blockDim.x + threadIdx.x;
  if (i < n) dst[i] = (half_t)src[i];
}

// src: K x N (row-major)  ->  dst: N x K (row-major) f16
__global__ void k_cvt16_t(const float* __restrict__ src, half_t* __restrict__ dst,
                          int K, int N) {
  int i = blockIdx.x * blockDim.x + threadIdx.x;
  if (i >= K * N) return;
  int k = i / N, n = i % N;
  dst[(size_t)n * K + k] = (half_t)src[i];
}

// c2_w (64,32,3,3) f32 -> w2t[oc][(dy*3+dx)*32+ic] f16 (64 x 288)
__global__ void k_reorder_c2w(const float* __restrict__ src, half_t* __restrict__ dst) {
  int i = blockIdx.x * blockDim.x + threadIdx.x;
  if (i >= 64 * 288) return;
  int oc = i / 288, r = i % 288;
  int dydx = r / 32, ic = r % 32;
  dst[i] = (half_t)src[(oc * 32 + ic) * 9 + dydx];
}

// ---------------------------------------------------------------------------
// Shared GEMM epilogue
// ---------------------------------------------------------------------------
static __device__ __forceinline__ void gemm_epilogue(
    v8f acc[2][2], int lane, int rowBase, int colBase,
    const float* bias, int biasOff, const float* gamma, const float* beta,
    const int* mask, int maskCol, float* out32, half_t* out16,
    int M, int N, int ldc, int cOff, int relu) {
  const int l15 = lane & 15;
#pragma unroll
  for (int i = 0; i < 2; ++i)
#pragma unroll
    for (int j = 0; j < 2; ++j)
#pragma unroll
      for (int r = 0; r < 8; ++r) {
        int row = rowBase + i * 16 + (lane >> 4) * 8 + r;
        int col = colBase + j * 16 + l15;
        if (row >= M || col >= N) continue;
        float v = acc[i][j][r];
        if (bias) v += bias[biasOff + col];
        if (gamma) v = v * (BN_INV * gamma[col]) + beta[col];
        if (relu) v = fmaxf(v, 0.0f);
        if (mask && maskCol >= 0) v *= (float)mask[row * 3 + maskCol];
        size_t o = (size_t)row * ldc + cOff + col;
        if (out32) out32[o] = v;
        if (out16) out16[o] = (half_t)v;
      }
}

// ---------------------------------------------------------------------------
// Generic WMMA GEMM (direct global loads):  C = epilogue(A[MxK] * Bt[NxK]^T)
// 128 threads = 4 waves; each wave a 32x32 tile; block tile 64x64.
// blockIdx.z batches experts.
// ---------------------------------------------------------------------------
__global__ __launch_bounds__(128) void k_gemm_wmma(
    const half_t* __restrict__ A, const half_t* __restrict__ Bt,
    const float* __restrict__ bias, const float* __restrict__ gamma,
    const float* __restrict__ beta, const int* __restrict__ mask, int maskCol,
    float* __restrict__ out32, half_t* __restrict__ out16,
    int M, int N, int K, int ldc, int colOff,
    int zBtStride, int zBiasN, int zColOff, int relu) {
  const int z = blockIdx.z;
  Bt += (size_t)z * zBtStride;
  const int biasOff = z * zBiasN;
  const int cOff = colOff + z * zColOff;

  const int lane = threadIdx.x & 31;
  const int wave = threadIdx.x >> 5;
  const int l15 = lane & 15;
  const int kb = (lane >> 4) * 8;
  const int rowBase = blockIdx.y * 64 + (wave >> 1) * 32;
  const int colBase = blockIdx.x * 64 + (wave & 1) * 32;

  v8f acc[2][2];
#pragma unroll
  for (int i = 0; i < 2; ++i)
#pragma unroll
    for (int j = 0; j < 2; ++j)
#pragma unroll
      for (int r = 0; r < 8; ++r) acc[i][j][r] = 0.0f;

  const half_t* aR0 = A + (size_t)(rowBase + l15) * K;
  const half_t* aR1 = A + (size_t)(rowBase + 16 + l15) * K;
  const half_t* bR0 = Bt + (size_t)(colBase + l15) * K;
  const half_t* bR1 = Bt + (size_t)(colBase + 16 + l15) * K;

  for (int k0 = 0; k0 < K; k0 += 32) {
    __builtin_prefetch(aR0 + k0 + 256, 0, 1);   // global_prefetch_b8
    __builtin_prefetch(bR0 + k0 + 256, 0, 1);
    v16h a0 = frag_load(aR0 + k0, kb);
    v16h a1 = frag_load(aR1 + k0, kb);
    v16h b0 = frag_load(bR0 + k0, kb);
    v16h b1 = frag_load(bR1 + k0, kb);
    acc[0][0] = wmma16(a0, b0, acc[0][0]);
    acc[0][1] = wmma16(a0, b1, acc[0][1]);
    acc[1][0] = wmma16(a1, b0, acc[1][0]);
    acc[1][1] = wmma16(a1, b1, acc[1][1]);
  }

  gemm_epilogue(acc, lane, rowBase, colBase, bias, biasOff, gamma, beta, mask,
                maskCol, out32, out16, M, N, ldc, cOff, relu);
}

// ---------------------------------------------------------------------------
// TDM-staged WMMA GEMM for large-K layers: the Tensor Data Mover streams
// double-buffered 64x64-half B-tiles into LDS (TENSORcnt fenced) while the
// waves run WMMA; A comes direct from global (compiler pipelines it).
// Requires K % 64 == 0, N % 64 == 0.
// ---------------------------------------------------------------------------
__global__ __launch_bounds__(128) void k_gemm_wmma_tdm(
    const half_t* __restrict__ A, const half_t* __restrict__ Bt,
    const float* __restrict__ bias, const float* __restrict__ gamma,
    const float* __restrict__ beta, const int* __restrict__ mask, int maskCol,
    float* __restrict__ out32, half_t* __restrict__ out16,
    int M, int N, int K, int ldc, int colOff, int relu) {
  __shared__ half_t sB[2][64 * 64];   // [buf][row*64 + k]

  const int lane = threadIdx.x & 31;
  const int wave = threadIdx.x >> 5;
  const int l15 = lane & 15;
  const int kb = (lane >> 4) * 8;
  const int rowBase = blockIdx.y * 64 + (wave >> 1) * 32;
  const int colBase0 = blockIdx.x * 64;          // block's B-row window
  const int colLoc = (wave & 1) * 32;            // wave's cols within tile

  v8f acc[2][2];
#pragma unroll
  for (int i = 0; i < 2; ++i)
#pragma unroll
    for (int j = 0; j < 2; ++j)
#pragma unroll
      for (int r = 0; r < 8; ++r) acc[i][j][r] = 0.0f;

  const half_t* aR0 = A + (size_t)(rowBase + l15) * K;
  const half_t* aR1 = A + (size_t)(rowBase + 16 + l15) * K;
  const half_t* btBase = Bt + (size_t)colBase0 * K;
  const int nChunks = K >> 6;

  if (wave == 0)
    tdm_load_tile_f16(btBase, (unsigned)(unsigned long long)&sB[0][0], 64u, 64u,
                      (unsigned)K);

  for (int c = 0; c < nChunks; ++c) {
    if (wave == 0) {
      if (c + 1 < nChunks) {
        tdm_load_tile_f16(btBase + (size_t)(c + 1) * 64,
                          (unsigned)(unsigned long long)&sB[(c + 1) & 1][0], 64u,
                          64u, (unsigned)K);
        __builtin_amdgcn_s_wait_tensorcnt(1);   // chunk c landed
      } else {
        __builtin_amdgcn_s_wait_tensorcnt(0);
      }
    }
    __syncthreads();   // sB[c&1] visible to all waves

    const half_t* sb = &sB[c & 1][0];
    const int k0 = c * 64;
#pragma unroll
    for (int s = 0; s < 2; ++s) {
      const int ks = s * 32;
      v16h a0 = frag_load(aR0 + k0 + ks, kb);
      v16h a1 = frag_load(aR1 + k0 + ks, kb);
      v16h b0 = frag_load(sb + (colLoc + l15) * 64 + ks, kb);
      v16h b1 = frag_load(sb + (colLoc + 16 + l15) * 64 + ks, kb);
      acc[0][0] = wmma16(a0, b0, acc[0][0]);
      acc[0][1] = wmma16(a0, b1, acc[0][1]);
      acc[1][0] = wmma16(a1, b0, acc[1][0]);
      acc[1][1] = wmma16(a1, b1, acc[1][1]);
    }
    __syncthreads();   // done reading before buffer is overwritten
  }

  gemm_epilogue(acc, lane, rowBase, colBase0 + colLoc, bias, 0, gamma, beta, mask,
                maskCol, out32, out16, M, N, ldc, colOff, relu);
}

// ---------------------------------------------------------------------------
// Fused image encoder: one workgroup (256 thr = 8 waves) per image.
// LDS: img f32 (4KB) | conv1 f16 [y][x][oc] (64KB) | pool1 f16 [p][ic] (16KB)
//      | conv2 f16 [p][oc] (32KB)  -> 118784 bytes dynamic smem.
// conv2 via implicit-GEMM WMMA: M=256 pixels, N=64 oc, K=288 (9 chunks of 32).
// ---------------------------------------------------------------------------
#define CONV_SMEM 118784
__global__ __launch_bounds__(256) void k_image_encoder(
    const float* __restrict__ img, const float* __restrict__ w1,
    const float* __restrict__ b1, const float* __restrict__ g1,
    const float* __restrict__ be1, const half_t* __restrict__ w2t,
    const float* __restrict__ b2, const float* __restrict__ g2,
    const float* __restrict__ be2, half_t* __restrict__ flat) {
  extern __shared__ char smem[];
  float* s_img = (float*)smem;                       // 1024 f32
  half_t* s_c1 = (half_t*)(smem + 4096);             // 32*32*32
  half_t* s_p1 = (half_t*)(smem + 4096 + 65536);     // 16*16*32
  half_t* s_c2 = (half_t*)(smem + 4096 + 65536 + 16384); // 16*16*64

  const int tid = threadIdx.x;
  const int b = blockIdx.x;

  for (int i = tid; i < 1024; i += 256) s_img[i] = img[(size_t)b * 1024 + i];
  __syncthreads();

  // conv1 + bias + BN + relu  -> s_c1[(y*32+x)*32+oc]
  for (int idx = tid; idx < 32768; idx += 256) {
    int oc = idx & 31, p = idx >> 5;
    int x = p & 31, y = p >> 5;
    float s = b1[oc];
#pragma unroll
    for (int dy = 0; dy < 3; ++dy) {
      int yy = y + dy - 1;
      if (yy < 0 || yy > 31) continue;
#pragma unroll
      for (int dx = 0; dx < 3; ++dx) {
        int xx = x + dx - 1;
        if (xx < 0 || xx > 31) continue;
        s += s_img[yy * 32 + xx] * w1[oc * 9 + dy * 3 + dx];
      }
    }
    s = s * (BN_INV * g1[oc]) + be1[oc];
    s_c1[idx] = (half_t)fmaxf(s, 0.0f);
  }
  __syncthreads();

  // maxpool 2x2 -> s_p1[(y*16+x)*32+ic]
  for (int idx = tid; idx < 8192; idx += 256) {
    int ic = idx & 31, p = idx >> 5;
    int x = p & 15, y = p >> 4;
    float m = (float)s_c1[((2 * y) * 32 + 2 * x) * 32 + ic];
    m = fmaxf(m, (float)s_c1[((2 * y) * 32 + 2 * x + 1) * 32 + ic]);
    m = fmaxf(m, (float)s_c1[((2 * y + 1) * 32 + 2 * x) * 32 + ic]);
    m = fmaxf(m, (float)s_c1[((2 * y + 1) * 32 + 2 * x + 1) * 32 + ic]);
    s_p1[idx] = (half_t)m;
  }
  __syncthreads();

  // conv2 implicit-GEMM WMMA: wave wv covers pixels [wv*32, wv*32+32), all 64 oc
  {
    const int lane = tid & 31;
    const int wv = tid >> 5;
    const int l15 = lane & 15;
    const int kb = (lane >> 4) * 8;
    v8f acc[2][4];
#pragma unroll
    for (int t = 0; t < 2; ++t)
#pragma unroll
      for (int j = 0; j < 4; ++j)
#pragma unroll
        for (int r = 0; r < 8; ++r) acc[t][j][r] = 0.0f;

    for (int ch = 0; ch < 9; ++ch) {
      int dy = ch / 3 - 1, dx = ch % 3 - 1;
      v16h a[2];
#pragma unroll
      for (int t = 0; t < 2; ++t) {
        int m = wv * 32 + t * 16 + l15;
        int y = (m >> 4) + dy, x = (m & 15) + dx;
        v16h av;
        if (y >= 0 && y < 16 && x >= 0 && x < 16) {
          const half_t* bp = s_p1 + (y * 16 + x) * 32;
          v8h lo = *(const v8h*)(bp + kb);
          v8h hi = *(const v8h*)(bp + 16 + kb);
#pragma unroll
          for (int i = 0; i < 8; ++i) { av[i] = lo[i]; av[8 + i] = hi[i]; }
        } else {
#pragma unroll
          for (int i = 0; i < 16; ++i) av[i] = (half_t)0.0f;
        }
        a[t] = av;
      }
#pragma unroll
      for (int j = 0; j < 4; ++j) {
        int n = j * 16 + l15;
        v16h bv = frag_load(w2t + n * 288 + ch * 32, kb);
        acc[0][j] = wmma16(a[0], bv, acc[0][j]);
        acc[1][j] = wmma16(a[1], bv, acc[1][j]);
      }
    }
    // bias + BN + relu -> s_c2[pixel*64 + oc]
#pragma unroll
    for (int t = 0; t < 2; ++t)
#pragma unroll
      for (int j = 0; j < 4; ++j)
#pragma unroll
        for (int r = 0; r < 8; ++r) {
          int pix = wv * 32 + t * 16 + (lane >> 4) * 8 + r;
          int oc = j * 16 + l15;
          float v = acc[t][j][r] + b2[oc];
          v = v * (BN_INV * g2[oc]) + be2[oc];
          s_c2[pix * 64 + oc] = (half_t)fmaxf(v, 0.0f);
        }
  }
  __syncthreads();

  // maxpool 2x2 + flatten NCHW order: flat[b, oc*64 + y*8 + x]
  for (int idx = tid; idx < 4096; idx += 256) {
    int oc = idx >> 6, p = idx & 63;
    int y = p >> 3, x = p & 7;
    float m = (float)s_c2[((2 * y) * 16 + 2 * x) * 64 + oc];
    m = fmaxf(m, (float)s_c2[((2 * y) * 16 + 2 * x + 1) * 64 + oc]);
    m = fmaxf(m, (float)s_c2[((2 * y + 1) * 16 + 2 * x) * 64 + oc]);
    m = fmaxf(m, (float)s_c2[((2 * y + 1) * 16 + 2 * x + 1) * 64 + oc]);
    flat[(size_t)b * 4096 + idx] = (half_t)m;
  }
}

// ---------------------------------------------------------------------------
// x_fused = fused_in + imp*(1-mask)   (fused_in already carries h*mask)
// ---------------------------------------------------------------------------
__global__ void k_fuse_impute(const float* __restrict__ fused32,
                              const float* __restrict__ imp32,
                              const int* __restrict__ mask,
                              float* __restrict__ xf32, half_t* __restrict__ xf16) {
  int i = blockIdx.x * blockDim.x + threadIdx.x;
  if (i >= B_SZ * FUS) return;
  int b = i / FUS, j = i % FUS;
  float m = (float)mask[b * 3 + (j / 256)];
  float v = fused32[i] + imp32[i] * (1.0f - m);
  xf32[i] = v;
  xf16[i] = (half_t)v;
}

// ---------------------------------------------------------------------------
// gate: conf = sigmoid(x_fused @ g_w + g_b), top-2  (one thread per row)
// ---------------------------------------------------------------------------
__global__ void k_gate_top2(const float* __restrict__ xf, const float* __restrict__ gw,
                            const float* __restrict__ gb, float* __restrict__ topv,
                            int* __restrict__ topi) {
  int b = blockIdx.x * blockDim.x + threadIdx.x;
  if (b >= B_SZ) return;
  float conf[NE];
#pragma unroll
  for (int e = 0; e < NE; ++e) conf[e] = gb[e];
  const float* row = xf + (size_t)b * FUS;
  for (int j = 0; j < FUS; ++j) {
    float xv = row[j];
#pragma unroll
    for (int e = 0; e < NE; ++e) conf[e] += xv * gw[j * NE + e];
  }
#pragma unroll
  for (int e = 0; e < NE; ++e) conf[e] = 1.0f / (1.0f + expf(-conf[e]));
  int i0 = 0; float v0 = conf[0];
#pragma unroll
  for (int e = 1; e < NE; ++e)
    if (conf[e] > v0) { v0 = conf[e]; i0 = e; }
  int i1 = -1; float v1 = -1e30f;
#pragma unroll
  for (int e = 0; e < NE; ++e)
    if (e != i0 && conf[e] > v1) { v1 = conf[e]; i1 = e; }
  topv[b * 2 + 0] = v0; topv[b * 2 + 1] = v1;
  topi[b * 2 + 0] = i0; topi[b * 2 + 1] = i1;
}

// ---------------------------------------------------------------------------
// final: out[b,c] = sum_t topv[t]*( he[b,idx_t] @ e_w2[idx_t][:,c] + e_b2 )
// ---------------------------------------------------------------------------
__global__ void k_expert_out(const half_t* __restrict__ he, const float* __restrict__ w2,
                             const float* __restrict__ b2, const float* __restrict__ topv,
                             const int* __restrict__ topi, float* __restrict__ out) {
  int i = blockIdx.x * blockDim.x + threadIdx.x;
  if (i >= B_SZ * NC) return;
  int b = i / NC, c = i % NC;
  float r = 0.0f;
#pragma unroll
  for (int t = 0; t < 2; ++t) {
    int e = topi[b * 2 + t];
    float s = b2[e * NC + c];
    const half_t* h = he + (size_t)b * (NE * 128) + e * 128;
    const float* w = w2 + (size_t)e * 128 * NC + c;
    for (int k = 0; k < 128; ++k) s += (float)h[k] * w[k * NC];
    r += topv[b * 2 + t] * s;
  }
  out[i] = r;
}

// ---------------------------------------------------------------------------
// Host: orchestrate on stream
// ---------------------------------------------------------------------------
extern "C" void kernel_launch(void* const* d_in, const int* in_sizes, int n_in,
                              void* d_out, int out_size, void* d_ws, size_t ws_size,
                              hipStream_t stream) {
  const float* x_csv = (const float*)d_in[0];
  const float* x_img1 = (const float*)d_in[1];
  const float* x_img2 = (const float*)d_in[2];
  const int* mask = (const int*)d_in[3];
  const float* s_w1 = (const float*)d_in[4];
  const float* s_b1 = (const float*)d_in[5];
  const float* s_g1 = (const float*)d_in[6];
  const float* s_be1 = (const float*)d_in[7];
  const float* s_w2 = (const float*)d_in[8];
  const float* s_b2 = (const float*)d_in[9];
  const float* s_g2 = (const float*)d_in[10];
  const float* s_be2 = (const float*)d_in[11];
  const float* c1_w = (const float*)d_in[12];
  const float* c1_b = (const float*)d_in[13];
  const float* c1_g = (const float*)d_in[14];
  const float* c1_be = (const float*)d_in[15];
  const float* c2_w = (const float*)d_in[16];
  const float* c2_b = (const float*)d_in[17];
  const float* c2_g = (const float*)d_in[18];
  const float* c2_be = (const float*)d_in[19];
  const float* fc_w = (const float*)d_in[20];
  const float* fc_b = (const float*)d_in[21];
  const float* fc_g = (const float*)d_in[22];
  const float* fc_be = (const float*)d_in[23];
  const float* i_w1 = (const float*)d_in[24];
  const float* i_b1 = (const float*)d_in[25];
  const float* i_w2 = (const float*)d_in[26];
  const float* i_b2 = (const float*)d_in[27];
  const float* g_w = (const float*)d_in[28];
  const float* g_b = (const float*)d_in[29];
  const float* e_w1 = (const float*)d_in[30];
  const float* e_b1 = (const float*)d_in[31];
  const float* e_w2 = (const float*)d_in[32];
  const float* e_b2 = (const float*)d_in[33];
  float* out = (float*)d_out;

  size_t off = 0;
  auto alloc = [&](size_t bytes) -> void* {
    void* p = (char*)d_ws + off;
    off += (bytes + 255) & ~(size_t)255;
    return p;
  };
  half_t* xcsv16 = (half_t*)alloc((size_t)B_SZ * CSV * 2);
  half_t* w1t = (half_t*)alloc((size_t)512 * CSV * 2);
  half_t* h1_16 = (half_t*)alloc((size_t)B_SZ * 512 * 2);
  half_t* w2t = (half_t*)alloc((size_t)EMB * 512 * 2);
  half_t* fct = (half_t*)alloc((size_t)EMB * 4096 * 2);
  half_t* flat1 = (half_t*)alloc((size_t)B_SZ * 4096 * 2);
  half_t* flat2 = (half_t*)alloc((size_t)B_SZ * 4096 * 2);
  half_t* c2wt = (half_t*)alloc((size_t)64 * 288 * 2);
  float* fused32 = (float*)alloc((size_t)B_SZ * FUS * 4);
  half_t* fused16 = (half_t*)alloc((size_t)B_SZ * FUS * 2);
  half_t* iw1t = (half_t*)alloc((size_t)512 * FUS * 2);
  half_t* imph16 = (half_t*)alloc((size_t)B_SZ * 512 * 2);
  half_t* iw2t = (half_t*)alloc((size_t)FUS * 512 * 2);
  float* imp32 = (float*)alloc((size_t)B_SZ * FUS * 4);
  float* xf32 = (float*)alloc((size_t)B_SZ * FUS * 4);
  half_t* xf16 = (half_t*)alloc((size_t)B_SZ * FUS * 2);
  half_t* ew1t = (half_t*)alloc((size_t)NE * 128 * FUS * 2);
  half_t* he16 = (half_t*)alloc((size_t)B_SZ * NE * 128 * 2);
  float* topv = (float*)alloc((size_t)B_SZ * 2 * 4);
  int* topi = (int*)alloc((size_t)B_SZ * 2 * 4);
  (void)ws_size; (void)n_in; (void)in_sizes; (void)out_size;

  auto g1d = [](int n) { return dim3((n + 255) / 256); };
  dim3 t256(256), t128(128);

  // --- precision conversion / weight transposes ---
  k_cvt16<<<g1d(B_SZ * CSV), t256, 0, stream>>>(x_csv, xcsv16, B_SZ * CSV);
  k_cvt16_t<<<g1d(CSV * 512), t256, 0, stream>>>(s_w1, w1t, CSV, 512);
  k_cvt16_t<<<g1d(512 * EMB), t256, 0, stream>>>(s_w2, w2t, 512, EMB);
  k_cvt16_t<<<g1d(4096 * EMB), t256, 0, stream>>>(fc_w, fct, 4096, EMB);
  k_cvt16_t<<<g1d(FUS * 512), t256, 0, stream>>>(i_w1, iw1t, FUS, 512);
  k_cvt16_t<<<g1d(512 * FUS), t256, 0, stream>>>(i_w2, iw2t, 512, FUS);
  k_reorder_c2w<<<g1d(64 * 288), t256, 0, stream>>>(c2_w, c2wt);
  for (int e = 0; e < NE; ++e)
    k_cvt16_t<<<g1d(FUS * 128), t256, 0, stream>>>(e_w1 + (size_t)e * FUS * 128,
                                                   ew1t + (size_t)e * 128 * FUS, FUS, 128);

  // --- sensor encoder (K=2048: TDM-staged B tiles) ---
  k_gemm_wmma_tdm<<<dim3(512 / 64, B_SZ / 64, 1), t128, 0, stream>>>(
      xcsv16, w1t, s_b1, s_g1, s_be1, nullptr, -1, nullptr, h1_16,
      B_SZ, 512, CSV, 512, 0, 1);
  k_gemm_wmma<<<dim3(EMB / 64, B_SZ / 64, 1), t128, 0, stream>>>(
      h1_16, w2t, s_b2, s_g2, s_be2, mask, 0, fused32, fused16,
      B_SZ, EMB, 512, FUS, 0, 0, 0, 0, 1);

  // --- image encoders (conv stack fused in LDS, conv2 via WMMA) ---
  k_image_encoder<<<dim3(B_SZ), t256, CONV_SMEM, stream>>>(
      x_img1, c1_w, c1_b, c1_g, c1_be, c2wt, c2_b, c2_g, c2_be, flat1);
  k_image_encoder<<<dim3(B_SZ), t256, CONV_SMEM, stream>>>(
      x_img2, c1_w, c1_b, c1_g, c1_be, c2wt, c2_b, c2_g, c2_be, flat2);
  // fc (K=4096): TDM-staged B tiles
  k_gemm_wmma_tdm<<<dim3(EMB / 64, B_SZ / 64, 1), t128, 0, stream>>>(
      flat1, fct, fc_b, fc_g, fc_be, mask, 1, fused32, fused16,
      B_SZ, EMB, 4096, FUS, 256, 1);
  k_gemm_wmma_tdm<<<dim3(EMB / 64, B_SZ / 64, 1), t128, 0, stream>>>(
      flat2, fct, fc_b, fc_g, fc_be, mask, 2, fused32, fused16,
      B_SZ, EMB, 4096, FUS, 512, 1);

  // --- imputation MLP ---
  k_gemm_wmma<<<dim3(512 / 64, B_SZ / 64, 1), t128, 0, stream>>>(
      fused16, iw1t, i_b1, nullptr, nullptr, nullptr, -1, nullptr, imph16,
      B_SZ, 512, FUS, 512, 0, 0, 0, 0, 1);
  k_gemm_wmma<<<dim3(FUS / 64, B_SZ / 64, 1), t128, 0, stream>>>(
      imph16, iw2t, i_b2, nullptr, nullptr, nullptr, -1, imp32, nullptr,
      B_SZ, FUS, 512, FUS, 0, 0, 0, 0, 0);
  k_fuse_impute<<<g1d(B_SZ * FUS), t256, 0, stream>>>(fused32, imp32, mask, xf32, xf16);

  // --- gate + top2 ---
  k_gate_top2<<<g1d(B_SZ), t256, 0, stream>>>(xf32, g_w, g_b, topv, topi);

  // --- experts (batched over blockIdx.z) ---
  k_gemm_wmma<<<dim3(128 / 64, B_SZ / 64, NE), t128, 0, stream>>>(
      xf16, ew1t, e_b1, nullptr, nullptr, nullptr, -1, nullptr, he16,
      B_SZ, 128, FUS, NE * 128, 0, 128 * FUS, 128, 128, 1);

  // --- gather + weighted sum ---
  k_expert_out<<<g1d(B_SZ * NC), t256, 0, stream>>>(he16, e_w2, e_b2, topv, topi, out);
}
